// NewRelativeMultiHeadAttention_35682588295638
// MI455X (gfx1250) — matrix operand order repaired and checked by reading, not compiled, not run on hardware
//
#include <hip/hip_runtime.h>

typedef __attribute__((ext_vector_type(16))) __bf16 v16bf;
typedef __attribute__((ext_vector_type(8)))  float  v8f;

#define BATCH   4
#define S_LEN   1024
#define DMODEL  512
#define N_HEAD  8
#define D_K     64
#define INV_TEMP 0.125f

__device__ __forceinline__ int lane_id() { return (int)(threadIdx.x & 31); }

__device__ __forceinline__ v8f zero8() {
  v8f z = {0.f, 0.f, 0.f, 0.f, 0.f, 0.f, 0.f, 0.f};
  return z;
}

// ---------------------------------------------------------------------------
// WMMA fragment loaders, per CDNA5 ISA 7.12.2 layouts (wave32).
// A (16x32 bf16, MxK): lane m (0-15) rows; lanes<16 hold K 0..7 & 16..23,
// lanes>=16 hold K 8..15 & 24..31.
// ---------------------------------------------------------------------------
__device__ __forceinline__ v16bf load_a_frag(const __bf16* p, int ld) {
  int l = lane_id();
  const __bf16* row = p + (size_t)(l & 15) * ld + ((l < 16) ? 0 : 8);
  v16bf a;
#pragma unroll
  for (int e = 0; e < 8; ++e) a[e] = row[e];
#pragma unroll
  for (int e = 0; e < 8; ++e) a[8 + e] = row[16 + e];
  return a;
}

// B (32x16 bf16, KxN) loaded from a ROW-MAJOR matrix W whose rows are the N
// dimension (B[k][n] = W[n][k]) -> contiguous 32B per lane.
// lanes<16: K 0..7 & 16..23, lanes>=16: K 8..15 & 24..31.
__device__ __forceinline__ v16bf load_b_fragT(const __bf16* p, int ld) {
  int l = lane_id();
  const __bf16* row = p + (size_t)(l & 15) * ld + ((l < 16) ? 0 : 8);
  v16bf b;
#pragma unroll
  for (int e = 0; e < 8; ++e) b[e] = row[e];
#pragma unroll
  for (int e = 0; e < 8; ++e) b[8 + e] = row[16 + e];
  return b;
}

__device__ __forceinline__ v8f wmma_bf16(v16bf a, v16bf b, v8f c) {
  return __builtin_amdgcn_wmma_f32_16x16x32_bf16(false, a, false, b, (short)0, c,
                                                 false, false);
}

// ---------------------------------------------------------------------------
// Kernel 0: f32 -> bf16 conversion, float4 vectorized
// ---------------------------------------------------------------------------
__global__ void cvt_bf16_kernel(const float4* __restrict__ src,
                                __bf16* __restrict__ dst, int n4) {
  int i = blockIdx.x * blockDim.x + threadIdx.x;
  if (i < n4) {
    float4 f = src[i];
    __bf16* d = dst + (size_t)i * 4;
    d[0] = (__bf16)f.x;
    d[1] = (__bf16)f.y;
    d[2] = (__bf16)f.z;
    d[3] = (__bf16)f.w;
  }
}

// ---------------------------------------------------------------------------
// Kernel 1: Weff[d][p] = sum_j rp_w2[d][j]*rp_w1[j][p]; beff = rp_w2@rp_b1+rp_b2
// weff layout: [0..63]=W0, [64..127]=W1, [128..191]=beff
// ---------------------------------------------------------------------------
__global__ void weff_kernel(const float* __restrict__ rp_w1,
                            const float* __restrict__ rp_b1,
                            const float* __restrict__ rp_w2,
                            const float* __restrict__ rp_b2,
                            float* __restrict__ weff) {
  int d = threadIdx.x;  // 64 threads
  float s0 = 0.f, s1 = 0.f, sb = 0.f;
  for (int j = 0; j < D_K; ++j) {
    float w2 = rp_w2[d * D_K + j];
    s0 += w2 * rp_w1[j * 2 + 0];
    s1 += w2 * rp_w1[j * 2 + 1];
    sb += w2 * rp_b1[j];
  }
  weff[d] = s0;
  weff[64 + d] = s1;
  weff[128 + d] = sb + rp_b2[d];
}

// ---------------------------------------------------------------------------
// Kernel 2: QKV projections. One wave per block, 16 rows x 64 cols (one head).
// grid = 3 * BATCH * (S/16) * N_HEAD = 6144.
// Q epilogue emits Qt0=Q*Weff0, Qt1=Q*Weff1, Qb=Q*beff (bf16, (bh,s,64)).
// K stored (bh,s,64); V stored transposed (bh,64,s) for contiguous B-frags.
// ---------------------------------------------------------------------------
__global__ void __launch_bounds__(32)
proj_kernel(const __bf16* __restrict__ qb, const __bf16* __restrict__ kb,
            const __bf16* __restrict__ vb, const __bf16* __restrict__ wq,
            const __bf16* __restrict__ wk, const __bf16* __restrict__ wv,
            const float* __restrict__ weff, __bf16* __restrict__ Qt0,
            __bf16* __restrict__ Qt1, __bf16* __restrict__ Qbv,
            __bf16* __restrict__ Kh, __bf16* __restrict__ Vt) {
  int id = blockIdx.x;
  int mat = id >> 11;       // 0=Q 1=K 2=V (2048 blocks each)
  int rem = id & 2047;
  int h = rem & 7;
  int stile = (rem >> 3) & 63;
  int b = rem >> 9;

  const __bf16* X = (mat == 0) ? qb : (mat == 1) ? kb : vb;
  const __bf16* W = (mat == 0) ? wq : (mat == 1) ? wk : wv;
  size_t row0 = (size_t)b * S_LEN + (size_t)stile * 16;

  v8f acc[4];
#pragma unroll
  for (int j = 0; j < 4; ++j) acc[j] = zero8();

  for (int ks = 0; ks < DMODEL / 32; ++ks) {
    v16bf a = load_a_frag(X + row0 * DMODEL + ks * 32, DMODEL);
#pragma unroll
    for (int j = 0; j < 4; ++j) {
      v16bf bw = load_b_fragT(W + (size_t)(h * 64 + j * 16) * DMODEL + ks * 32,
                              DMODEL);
      acc[j] = wmma_bf16(a, bw, acc[j]);
    }
  }

  int l = lane_id();
  int n = l & 15;
  int mb = (l < 16) ? 0 : 8;
  int bh = b * N_HEAD + h;

  if (mat == 0) {
#pragma unroll
    for (int j = 0; j < 4; ++j) {
      int d = j * 16 + n;
      float w0 = weff[d], w1 = weff[64 + d], wb = weff[128 + d];
#pragma unroll
      for (int r = 0; r < 8; ++r) {
        int srow = stile * 16 + mb + r;
        size_t idx = ((size_t)bh * S_LEN + srow) * D_K + d;
        float v = acc[j][r];
        Qt0[idx] = (__bf16)(v * w0);
        Qt1[idx] = (__bf16)(v * w1);
        Qbv[idx] = (__bf16)(v * wb);
      }
    }
  } else if (mat == 1) {
#pragma unroll
    for (int j = 0; j < 4; ++j) {
      int d = j * 16 + n;
#pragma unroll
      for (int r = 0; r < 8; ++r) {
        int srow = stile * 16 + mb + r;
        Kh[((size_t)bh * S_LEN + srow) * D_K + d] = (__bf16)acc[j][r];
      }
    }
  } else {
#pragma unroll
    for (int j = 0; j < 4; ++j) {
      int d = j * 16 + n;
#pragma unroll
      for (int r = 0; r < 8; ++r) {
        int srow = stile * 16 + mb + r;
        Vt[((size_t)bh * D_K + d) * S_LEN + srow] = (__bf16)acc[j][r];
      }
    }
  }
}

// ---------------------------------------------------------------------------
// Kernel 3: scores + pos combine + softmax + write attn (NT) + P@V.
// 4 waves per block cooperate on 16 q-rows of one (b,h):
//   scores : wave w owns k-tiles [w*16, w*16+16)  (disjoint LDS columns)
//   softmax: wave w owns rows    [w*4, w*4+4)
//   P@V    : wave w owns d-tile  j == w           (no cross-wave reduction)
// grid = 32 * 64 = 2048, 64 KB dynamic LDS -> 20 waves per WGP.
// ---------------------------------------------------------------------------
__global__ void __launch_bounds__(128)
attn_kernel(const __bf16* __restrict__ Qt0, const __bf16* __restrict__ Qt1,
            const __bf16* __restrict__ Qbv, const __bf16* __restrict__ Kh,
            const __bf16* __restrict__ Vt, const float* __restrict__ pos,
            float* __restrict__ attn_out, __bf16* __restrict__ Obf) {
  extern __shared__ float rowbuf[];  // [16][1024]
  int wave = (int)(threadIdx.x >> 5);
  int bh = blockIdx.x >> 6;
  int qblk = blockIdx.x & 63;
  int qbase = qblk * 16;
  int l = lane_id();
  int n = l & 15;
  int mb = (l < 16) ? 0 : 8;

  const __bf16* q0p = Qt0 + ((size_t)bh * S_LEN + qbase) * D_K;
  const __bf16* q1p = Qt1 + ((size_t)bh * S_LEN + qbase) * D_K;
  const __bf16* qbp = Qbv + ((size_t)bh * S_LEN + qbase) * D_K;

  v16bf a0[2], a1[2], ab[2];
#pragma unroll
  for (int s = 0; s < 2; ++s) {
    a0[s] = load_a_frag(q0p + s * 32, D_K);
    a1[s] = load_a_frag(q1p + s * 32, D_K);
    ab[s] = load_a_frag(qbp + s * 32, D_K);
  }

  const float2* pos2 = (const float2*)pos;

  for (int kt = wave * 16; kt < wave * 16 + 16; ++kt) {
    const __bf16* kp = Kh + ((size_t)bh * S_LEN + kt * 16) * D_K;
    v16bf bk0 = load_b_fragT(kp, D_K);
    v16bf bk1 = load_b_fragT(kp + 32, D_K);
    v8f s0 = zero8(), s1 = zero8(), sb = zero8();
    s0 = wmma_bf16(a0[0], bk0, s0);
    s0 = wmma_bf16(a0[1], bk1, s0);
    s1 = wmma_bf16(a1[0], bk0, s1);
    s1 = wmma_bf16(a1[1], bk1, s1);
    sb = wmma_bf16(ab[0], bk0, sb);
    sb = wmma_bf16(ab[1], bk1, sb);

    int k = kt * 16 + n;
#pragma unroll
    for (int r = 0; r < 8; ++r) {
      int qrow = qbase + mb + r;
      float2 pw = pos2[(size_t)qrow * S_LEN + k];  // L2-resident (8 MB)
      rowbuf[(mb + r) * S_LEN + k] =
          (pw.x * s0[r] + pw.y * s1[r] + sb[r]) * INV_TEMP;
    }
  }
  __syncthreads();

  // ---- softmax: 4 rows per wave; write attn output non-temporally ----
  float* arow0 = attn_out + ((size_t)bh * S_LEN + qbase) * S_LEN;
  for (int m = wave * 4; m < wave * 4 + 4; ++m) {
    float* row = rowbuf + m * S_LEN;
    float mx = -3.0e38f;
    for (int k = l; k < S_LEN; k += 32) mx = fmaxf(mx, row[k]);
#pragma unroll
    for (int off = 16; off > 0; off >>= 1)
      mx = fmaxf(mx, __shfl_xor(mx, off, 32));
    float sum = 0.f;
    for (int k = l; k < S_LEN; k += 32) {
      float e = __expf(row[k] - mx);
      row[k] = e;
      sum += e;
    }
#pragma unroll
    for (int off = 16; off > 0; off >>= 1) sum += __shfl_xor(sum, off, 32);
    float inv = 1.0f / sum;
    float* arow = arow0 + (size_t)m * S_LEN;
    for (int k = l; k < S_LEN; k += 32) {
      float pv = row[k] * inv;
      row[k] = pv;
      __builtin_nontemporal_store(pv, arow + k);  // 134 MB write-once stream
    }
  }
  __syncthreads();

  // ---- O(16x16) per wave = P(16x1024) @ V-slice, V stored transposed ----
  int j = wave;  // this wave's d-tile
  v8f o = zero8();
  for (int ks = 0; ks < S_LEN / 32; ++ks) {
    const float* prow =
        rowbuf + (size_t)(l & 15) * S_LEN + ks * 32 + ((l < 16) ? 0 : 8);
    v16bf a;
#pragma unroll
    for (int e = 0; e < 8; ++e) a[e] = (__bf16)prow[e];
#pragma unroll
    for (int e = 0; e < 8; ++e) a[8 + e] = (__bf16)prow[16 + e];
    v16bf bv =
        load_b_fragT(Vt + ((size_t)bh * D_K + j * 16) * S_LEN + ks * 32, S_LEN);
    o = wmma_bf16(a, bv, o);
  }

  int b = bh >> 3, h = bh & 7;
  int d = j * 16 + n;
#pragma unroll
  for (int r = 0; r < 8; ++r) {
    int srow = qbase + mb + r;
    Obf[((size_t)b * S_LEN + srow) * DMODEL + h * 64 + d] = (__bf16)o[r];
  }
}

// ---------------------------------------------------------------------------
// Kernel 4: out = O @ w_fc^T + residual, then LayerNorm. 4 waves per block,
// 16 rows x 512 cols: waves split nt-tiles (8 each), then LN rows (4 each).
// grid = 4096/16 = 256.
// ---------------------------------------------------------------------------
__global__ void __launch_bounds__(128)
fcln_kernel(const __bf16* __restrict__ Obf, const __bf16* __restrict__ wfc,
            const float* __restrict__ resid, const float* __restrict__ ln_g,
            const float* __restrict__ ln_b, float* __restrict__ out) {
  __shared__ __bf16 sA[16 * DMODEL];  // 16 KB
  __shared__ float sC[16 * DMODEL];   // 32 KB
  int wave = (int)(threadIdx.x >> 5);
  int l = lane_id();
  size_t row0 = (size_t)blockIdx.x * 16;

  for (int i = threadIdx.x; i < 16 * DMODEL; i += 128)
    sA[i] = Obf[row0 * DMODEL + i];
  __syncthreads();

  int n = l & 15;
  int mb = (l < 16) ? 0 : 8;
  for (int nt = wave * 8; nt < wave * 8 + 8; ++nt) {
    v8f acc = zero8();
    for (int ks = 0; ks < DMODEL / 32; ++ks) {
      v16bf a = load_a_frag(sA + ks * 32, DMODEL);
      v16bf bw =
          load_b_fragT(wfc + (size_t)(nt * 16) * DMODEL + ks * 32, DMODEL);
      acc = wmma_bf16(a, bw, acc);
    }
#pragma unroll
    for (int r = 0; r < 8; ++r) sC[(mb + r) * DMODEL + nt * 16 + n] = acc[r];
  }
  __syncthreads();

  for (int m = wave * 4; m < wave * 4 + 4; ++m) {
    size_t row = row0 + m;
    const float* rp = resid + row * DMODEL;
    float s = 0.f, s2 = 0.f;
    for (int c = l; c < DMODEL; c += 32) {
      float x = sC[m * DMODEL + c] + rp[c];
      sC[m * DMODEL + c] = x;
      s += x;
      s2 += x * x;
    }
#pragma unroll
    for (int off = 16; off > 0; off >>= 1) {
      s += __shfl_xor(s, off, 32);
      s2 += __shfl_xor(s2, off, 32);
    }
    float mu = s * (1.0f / DMODEL);
    float var = s2 * (1.0f / DMODEL) - mu * mu;
    float rs = rsqrtf(var + 1e-6f);
    float* op = out + row * DMODEL;
    for (int c = l; c < DMODEL; c += 32) {
      float y = (sC[m * DMODEL + c] - mu) * rs * ln_g[c] + ln_b[c];
      __builtin_nontemporal_store(y, op + c);
    }
  }
}

// ---------------------------------------------------------------------------
extern "C" void kernel_launch(void* const* d_in, const int* in_sizes, int n_in,
                              void* d_out, int out_size, void* d_ws,
                              size_t ws_size, hipStream_t stream) {
  (void)in_sizes; (void)n_in; (void)out_size; (void)ws_size;
  const float* q = (const float*)d_in[0];
  const float* k = (const float*)d_in[1];
  const float* v = (const float*)d_in[2];
  const float* pos_mat = (const float*)d_in[3];
  const float* w_qs = (const float*)d_in[4];
  const float* w_ks = (const float*)d_in[5];
  const float* w_vs = (const float*)d_in[6];
  const float* w_fc = (const float*)d_in[7];
  const float* rp_w1 = (const float*)d_in[8];
  const float* rp_b1 = (const float*)d_in[9];
  const float* rp_w2 = (const float*)d_in[10];
  const float* rp_b2 = (const float*)d_in[11];
  const float* ln_g = (const float*)d_in[12];
  const float* ln_b = (const float*)d_in[13];

  float* out = (float*)d_out;                              // (4,1024,512)
  float* attn_out = out + (size_t)BATCH * S_LEN * DMODEL;  // (4,8,1024,1024)

  const size_t NX = (size_t)BATCH * S_LEN * DMODEL;        // 2,097,152
  const size_t NW = (size_t)DMODEL * DMODEL;               // 262,144
  const size_t NQ = (size_t)BATCH * N_HEAD * S_LEN * D_K;  // 2,097,152

  __bf16* qb = (__bf16*)d_ws;
  __bf16* kb = qb + NX;
  __bf16* vb = kb + NX;
  __bf16* wqsb = vb + NX;
  __bf16* wksb = wqsb + NW;
  __bf16* wvsb = wksb + NW;
  __bf16* wfcb = wvsb + NW;
  __bf16* Qt0 = wfcb + NW;
  __bf16* Qt1 = Qt0 + NQ;
  __bf16* Qbv = Qt1 + NQ;
  __bf16* Kh = Qbv + NQ;
  __bf16* Vt = Kh + NQ;
  __bf16* Obf = Vt + NQ;
  float* weff = (float*)(Obf + NQ);  // 192 floats

  // Stage 0: conversions + effective pos-MLP weights
  cvt_bf16_kernel<<<(int)(NX / 4 / 256), 256, 0, stream>>>((const float4*)q,
                                                           qb, (int)(NX / 4));
  cvt_bf16_kernel<<<(int)(NX / 4 / 256), 256, 0, stream>>>((const float4*)k,
                                                           kb, (int)(NX / 4));
  cvt_bf16_kernel<<<(int)(NX / 4 / 256), 256, 0, stream>>>((const float4*)v,
                                                           vb, (int)(NX / 4));
  cvt_bf16_kernel<<<(int)(NW / 4 / 256), 256, 0, stream>>>(
      (const float4*)w_qs, wqsb, (int)(NW / 4));
  cvt_bf16_kernel<<<(int)(NW / 4 / 256), 256, 0, stream>>>(
      (const float4*)w_ks, wksb, (int)(NW / 4));
  cvt_bf16_kernel<<<(int)(NW / 4 / 256), 256, 0, stream>>>(
      (const float4*)w_vs, wvsb, (int)(NW / 4));
  cvt_bf16_kernel<<<(int)(NW / 4 / 256), 256, 0, stream>>>(
      (const float4*)w_fc, wfcb, (int)(NW / 4));
  weff_kernel<<<1, 64, 0, stream>>>(rp_w1, rp_b1, rp_w2, rp_b2, weff);

  // Stage 1: QKV projections (WMMA)
  proj_kernel<<<3 * 2048, 32, 0, stream>>>(qb, kb, vb, wqsb, wksb, wvsb, weff,
                                           Qt0, Qt1, Qbv, Kh, Vt);

  // Stage 2: scores + softmax + attn output + P@V (WMMA), 64 KB LDS, 4 waves
  attn_kernel<<<32 * 64, 128, 16 * S_LEN * sizeof(float), stream>>>(
      Qt0, Qt1, Qbv, Kh, Vt, pos_mat, attn_out, Obf);

  // Stage 3: fc + residual + LayerNorm (WMMA), 4 waves
  fcln_kernel<<<(BATCH * S_LEN) / 16, 128, 0, stream>>>(Obf, wfcb, q, ln_g,
                                                        ln_b, out);
}
